// Gemma4MoeRouter_26113401160075
// MI455X (gfx1250) — matrix operand order repaired and checked by reading, not compiled, not run on hardware
//
#include <hip/hip_runtime.h>
#include <hip/hip_bf16.h>

typedef __attribute__((ext_vector_type(16))) __bf16 v16bf;
typedef __attribute__((ext_vector_type(8)))  float  v8f;

#define H      2048
#define E      64
#define T      16384
#define TOPK   8
#define TPB    64                 // tokens per block in router_main / scatter
#define THREADS 128               // 4 waves
#define CHUNK  256
#define NCHUNK (H / CHUNK)        // 8
#define NBLK   (T / TPB)          // 256

union Frag { uint4 u[2]; v16bf v; };

// ---------------- kernel 0: W (f32) -> bf16 ----------------
__global__ void convert_w(const float* __restrict__ W, __bf16* __restrict__ wbf) {
  int i = blockIdx.x * 256 + threadIdx.x;
  wbf[i] = (__bf16)W[i];
}

// ---------------- kernel 1: fused RMSNorm + GEMM(WMMA bf16) + sigmoid + top-8 ----------------
__global__ __launch_bounds__(THREADS) void router_main(
    const float* __restrict__ x, const float* __restrict__ scale,
    const float* __restrict__ pes, const __bf16* __restrict__ wbf,
    int* __restrict__ experts8, float* __restrict__ scores8,
    int* __restrict__ partial)
{
  extern __shared__ char smem[];
  __bf16* yLds   = (__bf16*)smem;                 // TPB x CHUNK bf16 = 32768 B
  float*  scores = (float*)(smem + 32768);        // TPB x E f32    = 16384 B
  float*  sumsP  = (float*)(smem + 49152);        // TPB x 2        = 512 B
  float*  rsArr  = (float*)(smem + 49664);        // TPB            = 256 B
  float*  pesL   = (float*)(smem + 49920);        // E              = 256 B
  int*    hist   = (int*)  (smem + 50176);        // E              = 256 B

  const int tid   = threadIdx.x;
  const int lane  = tid & 31;
  const int wave  = tid >> 5;            // 0..3, 16 tokens each
  const int l15   = lane & 15;
  const int lhalf = lane >> 4;           // 0 or 1
  const int t0    = blockIdx.x * TPB;

  if (tid < E)     { pesL[tid] = pes[tid]; hist[tid] = 0; }
  if (tid < TPB*2) sumsP[tid] = 0.0f;
  __syncthreads();

  v8f acc[4];
  #pragma unroll
  for (int j = 0; j < 4; ++j)
    #pragma unroll
    for (int r = 0; r < 8; ++r) acc[j][r] = 0.0f;

  for (int c = 0; c < NCHUNK; ++c) {
    const int kbase = c * CHUNK;
    // ---- stage chunk: yLds = bf16(x*scale); accumulate sum(x^2) deterministically ----
    #pragma unroll 4
    for (int it = 0; it < 32; ++it) {
      int flat = it * (THREADS*4) + tid * 4;          // 0..16383
      int trow = flat >> 8;                           // /CHUNK, wave-uniform
      int col  = flat & (CHUNK - 1);
      const float4 xv = *(const float4*)(x + (size_t)(t0 + trow)*H + kbase + col);
      const float4 sv = *(const float4*)(scale + kbase + col);
      __bf16* dst = yLds + trow*CHUNK + col;
      dst[0] = (__bf16)(xv.x*sv.x); dst[1] = (__bf16)(xv.y*sv.y);
      dst[2] = (__bf16)(xv.z*sv.z); dst[3] = (__bf16)(xv.w*sv.w);
      float ss = xv.x*xv.x + xv.y*xv.y + xv.z*xv.z + xv.w*xv.w;
      #pragma unroll
      for (int o = 16; o > 0; o >>= 1) ss += __shfl_xor(ss, o, 32);
      if (lane == 0) sumsP[trow*2 + (wave & 1)] += ss;   // unique (token,half) owner
    }
    __syncthreads();
    // ---- WMMA over the chunk: 16 tokens x 64 experts per wave ----
    #pragma unroll
    for (int ks = 0; ks < CHUNK/32; ++ks) {
      const int kb = ks * 32;
      // A 16x32 bf16: lane<16 holds K {kb..kb+7, kb+16..kb+23}; lane>=16 shifted by 8
      const __bf16* ap = yLds + (wave*16 + l15)*CHUNK + kb + lhalf*8;
      Frag A; A.u[0] = *(const uint4*)ap; A.u[1] = *(const uint4*)(ap + 16);
      #pragma unroll
      for (int j = 0; j < 4; ++j) {
        // B 32x16 bf16: column = expert j*16+l15; lane<16 holds K {kb..kb+7, kb+16..kb+23}; lane>=16 shifted by 8
        const __bf16* bp = wbf + (size_t)(j*16 + l15)*H + kbase + kb + lhalf*8;
        Frag B; B.u[0] = *(const uint4*)bp; B.u[1] = *(const uint4*)(bp + 16);
        acc[j] = __builtin_amdgcn_wmma_f32_16x16x32_bf16(
            false, A.v, false, B.v, (short)0, acc[j], false, false);
      }
    }
    __syncthreads();
  }

  if (tid < TPB)
    rsArr[tid] = rsqrtf((sumsP[tid*2] + sumsP[tid*2+1]) * (1.0f/(float)H) + 1e-6f);
  __syncthreads();

  // logits * rsqrt * per_expert_scale -> sigmoid -> LDS scores
  #pragma unroll
  for (int j = 0; j < 4; ++j)
    #pragma unroll
    for (int r = 0; r < 8; ++r) {
      int m = wave*16 + r + lhalf*8;     // C layout: VGPR r, lane half selects M / M+8
      int e = j*16 + l15;
      float v = acc[j][r] * rsArr[m] * pesL[e];
      scores[m*E + e] = 1.0f / (1.0f + __expf(-v));
    }
  __syncthreads();

  // stable top-8 per token (strict > keeps lowest index on ties)
  if (tid < TPB) {
    float* row = scores + tid*E;
    int gt = t0 + tid;
    for (int k = 0; k < TOPK; ++k) {
      float best = -1.0f; int bi = 0;
      for (int e = 0; e < E; ++e) { float s = row[e]; if (s > best) { best = s; bi = e; } }
      row[bi] = -2.0f;
      experts8[gt*TOPK + k] = bi;
      scores8 [gt*TOPK + k] = best;
      atomicAdd(&hist[bi], 1);
    }
  }
  __syncthreads();
  if (tid < E) partial[blockIdx.x*E + tid] = hist[tid];
}

// ---------------- kernel 2: per-expert scan of per-block histograms ----------------
__global__ void scan_kernel(const int* __restrict__ partial, int* __restrict__ blockBase,
                            float* __restrict__ outCounts) {
  __shared__ int totals[E];
  __shared__ int base[E];
  int e = threadIdx.x;                  // 0..63
  int tot = 0;
  for (int b = 0; b < NBLK; ++b) tot += partial[b*E + e];
  totals[e] = tot;
  outCounts[e] = (float)tot;            // num_tokens_per_expert
  __syncthreads();
  if (e == 0) { int run = 0; for (int i = 0; i < E; ++i) { base[i] = run; run += totals[i]; } }
  __syncthreads();
  int run = base[e];
  for (int b = 0; b < NBLK; ++b) { blockBase[b*E + e] = run; run += partial[b*E + e]; }
}

// ---------------- kernel 3: stable scatter (counting sort by expert) ----------------
__global__ __launch_bounds__(TPB) void scatter_kernel(
    const int* __restrict__ experts8, const float* __restrict__ scores8,
    const int* __restrict__ blockBase, float* __restrict__ outScores,
    float* __restrict__ outIdx)
{
  __shared__ unsigned mLo[TPB], mHi[TPB];
  __shared__ int baseRow[E];
  const int t = threadIdx.x, b = blockIdx.x;
  const int gt = b*TPB + t;
  int ex[TOPK]; float sc[TOPK];
  unsigned lo = 0, hi = 0;
  #pragma unroll
  for (int k = 0; k < TOPK; ++k) {
    ex[k] = experts8[gt*TOPK + k];
    sc[k] = scores8 [gt*TOPK + k];
    if (ex[k] < 32) lo |= 1u << ex[k]; else hi |= 1u << (ex[k] - 32);
  }
  mLo[t] = lo; mHi[t] = hi;
  baseRow[t] = blockBase[b*E + t];      // TPB == E == 64
  __syncthreads();
  int rank[TOPK] = {0,0,0,0,0,0,0,0};
  for (int tp = 0; tp < t; ++tp) {      // experts distinct within a token => rank = #earlier tokens
    unsigned plo = mLo[tp], phi = mHi[tp];
    #pragma unroll
    for (int k = 0; k < TOPK; ++k) {
      int e = ex[k];
      rank[k] += (int)((e < 32 ? (plo >> e) : (phi >> (e - 32))) & 1u);
    }
  }
  #pragma unroll
  for (int k = 0; k < TOPK; ++k) {
    int p = baseRow[ex[k]] + rank[k];
    outScores[p] = sc[k];
    outIdx[p]    = (float)gt;
  }
}

extern "C" void kernel_launch(void* const* d_in, const int* in_sizes, int n_in,
                              void* d_out, int out_size, void* d_ws, size_t ws_size,
                              hipStream_t stream) {
  const float* x     = (const float*)d_in[0];
  const float* scale = (const float*)d_in[1];
  const float* W     = (const float*)d_in[2];
  const float* pes   = (const float*)d_in[3];
  float* out = (float*)d_out;
  char*  ws  = (char*)d_ws;

  // workspace layout
  __bf16* wbf       = (__bf16*)(ws);                       // 64*2048*2    = 262144 B
  int*    experts8  = (int*)   (ws + 262144);              // 131072*4     = 524288 B
  float*  scores8   = (float*) (ws + 262144 + 524288);     // 131072*4     = 524288 B
  int*    partial   = (int*)   (ws + 1310720);             // 256*64*4     = 65536 B
  int*    blockBase = (int*)   (ws + 1310720 + 65536);     // 256*64*4     = 65536 B

  convert_w  <<<(E*H)/256, 256, 0, stream>>>(W, wbf);
  router_main<<<NBLK, THREADS, 50432, stream>>>(x, scale, pes, wbf, experts8, scores8, partial);
  scan_kernel<<<1, E, 0, stream>>>(partial, blockBase, out + 2*T*TOPK);
  scatter_kernel<<<NBLK, TPB, 0, stream>>>(experts8, scores8, blockBase, out, out + T*TOPK);
}
